// TreeLSTM_6605659702093
// MI455X (gfx1250) — matrix operand order repaired and checked by reading, not compiled, not run on hardware
//
#include <hip/hip_runtime.h>
#include <hip/hip_bf16.h>

typedef __attribute__((ext_vector_type(16))) __bf16 v16bf;
typedef __attribute__((ext_vector_type(8)))  __bf16 v8bf;
typedef __attribute__((ext_vector_type(8)))  float  v8f;

#define BSZ   512
#define NPTN  1023
#define XDIM  128
#define HDIM  64
#define DEPTHC 9
#define CCLS  16

__device__ __forceinline__ float sigf(float x) { return 1.0f / (1.0f + __expf(-x)); }

// ---------------------------------------------------------------------------
// Repack weights into WMMA B-fragment layout (bf16).
// Fragment (nt, kt) covers B[k = kt*32 .. +31][n = nt*16 .. +15].
// Per ISA 7.12: lane L holds column n = L&15, K = 8*(L>>4) + (e&7) + 16*(e>>3) (e=0..15).
// Fragment storage: frag_id*512 + L*16 + e  (each lane reads 32B contiguous).
// wW: W_iou [128 x 192]  -> 12 nt * 4 kt frags (24576 elems)
// wU: [U_iou | Uf_w^T] [128 x 320] -> 20 nt * 4 kt frags (40960 elems)
// ---------------------------------------------------------------------------
__global__ void prep_weights(const float* __restrict__ W_iou,
                             const float* __restrict__ U_iou,
                             const float* __restrict__ Uf_w,
                             __bf16* __restrict__ wW, __bf16* __restrict__ wU) {
  int idx = blockIdx.x * blockDim.x + threadIdx.x;
  if (idx < 24576) {
    int f = idx >> 9, rem = idx & 511;
    int L = rem >> 4, e = rem & 15;
    int nt = f >> 2, kt = f & 3;
    int k = kt * 32 + ((L >> 4) << 3) + (e & 7) + ((e >> 3) << 4);
    int n = nt * 16 + (L & 15);
    wW[idx] = (__bf16)W_iou[k * 192 + n];
  }
  if (idx < 40960) {
    int f = idx >> 9, rem = idx & 511;
    int L = rem >> 4, e = rem & 15;
    int nt = f >> 2, kt = f & 3;
    int k = kt * 32 + ((L >> 4) << 3) + (e & 7) + ((e >> 3) << 4);
    int nc = nt * 16 + (L & 15);
    float v = (nc < 192) ? U_iou[k * 192 + nc]        // iou block
                         : Uf_w[(nc - 192) * 128 + k]; // f block = Uf_w^T
    wU[idx] = (__bf16)v;
  }
}

// ---------------------------------------------------------------------------
// Leaf kernel: rows = B*512 leaves. iou = x @ W_iou + b_iou; gates with c0.
// Block: 256 thr = 8 waves, 32 rows, N=192 (12 n-tiles; 3 per wave).
// ---------------------------------------------------------------------------
__global__ void leaf_kernel(const float* __restrict__ emb,
                            const float* __restrict__ c0,
                            const __bf16* __restrict__ wW,
                            const float* __restrict__ b_iou,
                            __bf16* __restrict__ h, float* __restrict__ c) {
  __shared__ float Gs[32 * 192];
  const int tid = threadIdx.x;
  const int wave = tid >> 5, lane = tid & 31;
  const int m_sub = wave & 1, n_grp = wave >> 1;   // n_grp 0..3
  const int half = lane >> 4, m = lane & 15;
  const int row0 = blockIdx.x * 32;

  // A-operand row (per-lane): embeddings row for this leaf
  {
    const int r = row0 + m_sub * 16 + m;
    const int b = r >> 9;
    const int node = 511 + (r & 511);
    const float* arow = emb + (size_t)(b * NPTN + node) * 128;

    v8f acc[3] = {};
#pragma unroll
    for (int kt = 0; kt < 4; ++kt) {
      const int k0 = kt * 32 + half * 8;
      float4 a0 = *(const float4*)(arow + k0);
      float4 a1 = *(const float4*)(arow + k0 + 4);
      float4 a2 = *(const float4*)(arow + k0 + 16);
      float4 a3 = *(const float4*)(arow + k0 + 20);
      v16bf af;
      af[0]=(__bf16)a0.x;  af[1]=(__bf16)a0.y;  af[2]=(__bf16)a0.z;  af[3]=(__bf16)a0.w;
      af[4]=(__bf16)a1.x;  af[5]=(__bf16)a1.y;  af[6]=(__bf16)a1.z;  af[7]=(__bf16)a1.w;
      af[8]=(__bf16)a2.x;  af[9]=(__bf16)a2.y;  af[10]=(__bf16)a2.z; af[11]=(__bf16)a2.w;
      af[12]=(__bf16)a3.x; af[13]=(__bf16)a3.y; af[14]=(__bf16)a3.z; af[15]=(__bf16)a3.w;
#pragma unroll
      for (int i = 0; i < 3; ++i) {
        const int nt = n_grp * 3 + i;
        v16bf bfrag = *(const v16bf*)(wW + (size_t)((nt * 4 + kt) * 512 + lane * 16));
        acc[i] = __builtin_amdgcn_wmma_f32_16x16x32_bf16(
            false, af, false, bfrag, (short)0, acc[i], false, false);
      }
    }
#pragma unroll
    for (int i = 0; i < 3; ++i) {
      const int nt = n_grp * 3 + i;
#pragma unroll
      for (int v = 0; v < 8; ++v) {
        const int rl = m_sub * 16 + v + half * 8;   // D layout: row = v + 8*(lane>>4)
        Gs[rl * 192 + nt * 16 + m] = acc[i][v];
      }
    }
  }
  __syncthreads();

  // Epilogue: gates
  const int q = tid & 63;
  const float bi = b_iou[q], bo = b_iou[64 + q], bu = b_iou[128 + q];
#pragma unroll
  for (int it = 0; it < 8; ++it) {
    const int rl = (tid >> 6) + it * 4;
    const int rr = row0 + rl;
    const int bb = rr >> 9;
    const int nd = 511 + (rr & 511);
    const size_t off = (size_t)(bb * NPTN + nd) * 64 + q;
    const float i_ = Gs[rl * 192 + q] + bi;
    const float o_ = Gs[rl * 192 + 64 + q] + bo;
    const float u_ = Gs[rl * 192 + 128 + q] + bu;
    const float cn = sigf(i_) * tanhf(u_) + c0[off];
    h[off] = (__bf16)(sigf(o_) * tanhf(cn));
    c[off] = cn;
  }
}

// ---------------------------------------------------------------------------
// Internal-level kernel. Rows = B * n nodes at level lvl (n = 2^lvl).
// hc = children h (contiguous 128 bf16!). GEMM vs fused [U_iou | Uf_w^T].
// Block: 256 thr = 8 waves, 32 rows, N=320 (20 n-tiles; 5 per wave).
// ---------------------------------------------------------------------------
__global__ void level_kernel(const __bf16* __restrict__ wU,
                             const float* __restrict__ b_iou,
                             const float* __restrict__ Uf_b,
                             __bf16* __restrict__ h, float* __restrict__ c,
                             int s0, int n, int lg) {
  __shared__ float Gs[32 * 320];
  const int tid = threadIdx.x;
  const int wave = tid >> 5, lane = tid & 31;
  const int m_sub = wave & 1, n_grp = wave >> 1;   // n_grp 0..3
  const int half = lane >> 4, m = lane & 15;
  const int row0 = blockIdx.x * 32;

  {
    const int r = row0 + m_sub * 16 + m;
    const int b = r >> lg;
    const int j = r & (n - 1);
    const int node = s0 + j;
    // children (2*node+1, 2*node+2) are adjacent -> hc = 128 contiguous bf16
    const __bf16* arow = h + (size_t)(b * NPTN + 2 * node + 1) * 64;

    v8f acc[5] = {};
#pragma unroll
    for (int kt = 0; kt < 4; ++kt) {
      const int k0 = kt * 32 + half * 8;
      v8bf c1 = *(const v8bf*)(arow + k0);        // K = k0..k0+7
      v8bf c2 = *(const v8bf*)(arow + k0 + 16);   // K = k0+16..k0+23
      v16bf af;
#pragma unroll
      for (int e = 0; e < 8; ++e) { af[e] = c1[e]; af[8 + e] = c2[e]; }
#pragma unroll
      for (int i = 0; i < 5; ++i) {
        const int nt = n_grp * 5 + i;
        v16bf bfrag = *(const v16bf*)(wU + (size_t)((nt * 4 + kt) * 512 + lane * 16));
        acc[i] = __builtin_amdgcn_wmma_f32_16x16x32_bf16(
            false, af, false, bfrag, (short)0, acc[i], false, false);
      }
    }
#pragma unroll
    for (int i = 0; i < 5; ++i) {
      const int nt = n_grp * 5 + i;
#pragma unroll
      for (int v = 0; v < 8; ++v) {
        const int rl = m_sub * 16 + v + half * 8;
        Gs[rl * 320 + nt * 16 + m] = acc[i][v];
      }
    }
  }
  __syncthreads();

  // Epilogue: f-gated child reduction + iou gates
  const int q = tid & 63;
  const float bi = b_iou[q], bo = b_iou[64 + q], bu = b_iou[128 + q];
  const float bfl = Uf_b[q], bfr = Uf_b[64 + q];
#pragma unroll
  for (int it = 0; it < 8; ++it) {
    const int rl = (tid >> 6) + it * 4;
    const int rr = row0 + rl;
    const int bb = rr >> lg;
    const int jj = rr & (n - 1);
    const int nd = s0 + jj;
    const size_t cb = (size_t)(bb * NPTN + 2 * nd + 1) * 64;  // children c base
    const float fl = sigf(Gs[rl * 320 + 192 + q] + bfl);
    const float fr = sigf(Gs[rl * 320 + 256 + q] + bfr);
    const float cred = fl * c[cb + q] + fr * c[cb + 64 + q];
    const float i_ = Gs[rl * 320 + q] + bi;
    const float o_ = Gs[rl * 320 + 64 + q] + bo;
    const float u_ = Gs[rl * 320 + 128 + q] + bu;
    const float cn = sigf(i_) * tanhf(u_) + cred;
    const size_t off = (size_t)(bb * NPTN + nd) * 64 + q;
    h[off] = (__bf16)(sigf(o_) * tanhf(cn));
    c[off] = cn;
  }
}

// ---------------------------------------------------------------------------
// Mean pool over 1023 nodes, classifier [64x16], softmax. One block per tree.
// ---------------------------------------------------------------------------
__global__ void final_kernel(const __bf16* __restrict__ h,
                             const float* __restrict__ lin_w,
                             const float* __restrict__ lin_b,
                             float* __restrict__ out) {
  __shared__ float red[4][64];
  __shared__ float hm[64];
  __shared__ float lgt[16];
  __shared__ float ex[16];
  const int b = blockIdx.x, tid = threadIdx.x;
  const int q = tid & 63, g = tid >> 6;
  float s = 0.f;
  for (int node = g; node < NPTN; node += 4)
    s += (float)h[(size_t)(b * NPTN + node) * 64 + q];
  red[g][q] = s;
  __syncthreads();
  if (tid < 64)
    hm[q] = (red[0][q] + red[1][q] + red[2][q] + red[3][q]) * (1.0f / 1023.0f);
  __syncthreads();
  if (tid < 16) {
    float l = lin_b[tid];
#pragma unroll
    for (int k = 0; k < 64; ++k) l += hm[k] * lin_w[k * 16 + tid];
    lgt[tid] = l;
  }
  __syncthreads();
  if (tid < 16) {
    float mx = lgt[0];
#pragma unroll
    for (int k = 1; k < 16; ++k) mx = fmaxf(mx, lgt[k]);
    ex[tid] = __expf(lgt[tid] - mx);
  }
  __syncthreads();
  if (tid < 16) {
    float sm = 0.f;
#pragma unroll
    for (int k = 0; k < 16; ++k) sm += ex[k];
    out[b * 16 + tid] = ex[tid] / sm;
  }
}

// ---------------------------------------------------------------------------
extern "C" void kernel_launch(void* const* d_in, const int* in_sizes, int n_in,
                              void* d_out, int out_size, void* d_ws, size_t ws_size,
                              hipStream_t stream) {
  const float* emb   = (const float*)d_in[0];
  // d_in[1] = h0: unused (every node's h is overwritten)
  const float* c0    = (const float*)d_in[2];
  const float* W_iou = (const float*)d_in[3];
  const float* U_iou = (const float*)d_in[4];
  const float* b_iou = (const float*)d_in[5];
  const float* Uf_w  = (const float*)d_in[6];
  const float* Uf_b  = (const float*)d_in[7];
  const float* lin_w = (const float*)d_in[8];
  const float* lin_b = (const float*)d_in[9];
  float* out = (float*)d_out;

  const size_t BNH = (size_t)BSZ * NPTN * HDIM;   // 33,521,664
  __bf16* h  = (__bf16*)d_ws;                     // BNH bf16  (~67 MB)
  float*  c  = (float*)((char*)d_ws + BNH * sizeof(__bf16));   // BNH f32 (~134 MB)
  __bf16* wW = (__bf16*)((char*)c + BNH * sizeof(float));      // 24576 bf16
  __bf16* wU = wW + 24576;                                     // 40960 bf16

  prep_weights<<<160, 256, 0, stream>>>(W_iou, U_iou, Uf_w, wW, wU);
  leaf_kernel<<<(BSZ * 512) / 32, 256, 0, stream>>>(emb, c0, wW, b_iou, h, c);
  for (int lvl = DEPTHC - 1; lvl >= 0; --lvl) {
    const int nn = 1 << lvl;
    level_kernel<<<(BSZ * nn) / 32, 256, 0, stream>>>(wU, b_iou, Uf_b, h, c,
                                                      nn - 1, nn, lvl);
  }
  final_kernel<<<BSZ, 256, 0, stream>>>(h, lin_w, lin_b, out);
}
